// CausalSelfAttention_48473000903203
// MI455X (gfx1250) — matrix operand order repaired and checked by reading, not compiled, not run on hardware
//
#include <hip/hip_runtime.h>

// CausalSelfAttention forward for MI455X (gfx1250): bf16 WMMA + TDM-staged LDS.
// Workspace usage (bytes): xb 16MiB + wqkvT 24MiB + wprojT 8MiB + qkv 48MiB + y 16MiB = 112MiB.

typedef __bf16 bf16_t;
typedef __attribute__((ext_vector_type(16))) __bf16 v16bf;
typedef __attribute__((ext_vector_type(8)))  __bf16 v8bf;
typedef __attribute__((ext_vector_type(8)))  float  v8f;
typedef __attribute__((ext_vector_type(4)))  unsigned int u32x4;
typedef __attribute__((ext_vector_type(8)))  int i32x8;
typedef __attribute__((ext_vector_type(4)))  int i32x4;

#define D_MODEL 2048
#define N_HEAD  16
#define HEAD_DIM 128
#define SEQ_T   1024
#define BATCH   4
#define NROW    (BATCH * SEQ_T)   // 4096
#define QKV_N   (3 * D_MODEL)     // 6144

static __device__ __forceinline__ v16bf pack16(v8bf lo, v8bf hi) {
  v16bf r;
#pragma unroll
  for (int i = 0; i < 8; ++i) { r[i] = lo[i]; r[i + 8] = hi[i]; }
  return r;
}

// Low 32 bits of a generic pointer to __shared__ = LDS byte offset (aperture model).
static __device__ __forceinline__ unsigned lds_off(const void* p) {
  return (unsigned)(unsigned long long)(uintptr_t)p;
}

// Issue a TDM 2D tile load: tile_d0 contiguous elements (bf16) per row,
// tile_d1 rows at stride0 elements; packed contiguously into LDS at lds_addr.
// Wave-level DMA, tracked with TENSORcnt. Descriptor per CDNA5 ISA ch.8.
static __device__ __forceinline__ void tdm_load_2d(unsigned lds_addr, const void* gptr,
                                                   unsigned tile_d0, unsigned tile_d1,
                                                   unsigned long long stride0_elems) {
  const unsigned long long ga = (unsigned long long)(uintptr_t)gptr;
  u32x4 g0;
  g0[0] = 1u;                                    // count=1, user desc
  g0[1] = lds_addr;                              // LDS byte address
  g0[2] = (unsigned)ga;                          // global_addr[31:0]
  g0[3] = (unsigned)(ga >> 32) | (2u << 30);     // global_addr[56:32] | type=2
  i32x8 g1;
  g1[0] = (int)(1u << 16);                                            // data_size=1 (2B)
  g1[1] = (int)((tile_d0 & 0xFFFFu) << 16);                           // tensor_dim0[15:0]
  g1[2] = (int)((tile_d0 >> 16) | ((tile_d1 & 0xFFFFu) << 16));       // d0 hi | tensor_dim1 lo
  g1[3] = (int)((tile_d1 >> 16) | ((tile_d0 & 0xFFFFu) << 16));       // d1 hi | tile_dim0
  g1[4] = (int)(tile_d1 & 0xFFFFu);                                   // tile_dim1 (tile_dim2=0)
  g1[5] = (int)(unsigned)(stride0_elems & 0xFFFFFFFFull);             // tensor_dim0_stride lo
  g1[6] = (int)(unsigned)((stride0_elems >> 32) & 0xFFFFull);         // stride hi (dim1_stride=0)
  g1[7] = 0;
  i32x4 z4 = {};
#if __clang_major__ >= 23
  i32x8 z8 = {};
  __builtin_amdgcn_tensor_load_to_lds(g0, g1, z4, z4, z8, 0);
#else
  __builtin_amdgcn_tensor_load_to_lds(g0, g1, z4, z4, 0);
#endif
}

// ---------------------------------------------------------------------------
// 1) fp32 -> bf16 elementwise convert (n divisible by 4)
// ---------------------------------------------------------------------------
__global__ void cvt_f32_bf16(const float* __restrict__ in, bf16_t* __restrict__ out, int n) {
  int i = (blockIdx.x * blockDim.x + threadIdx.x) * 4;
  if (i + 3 < n) {
    float4 v = *(const float4*)(in + i);
    out[i + 0] = (bf16_t)v.x;
    out[i + 1] = (bf16_t)v.y;
    out[i + 2] = (bf16_t)v.z;
    out[i + 3] = (bf16_t)v.w;
  }
}

// ---------------------------------------------------------------------------
// 2) transpose + convert: in fp32 [K,N] row-major -> out bf16 [N,K] row-major
// ---------------------------------------------------------------------------
__global__ void transpose_cvt(const float* __restrict__ in, bf16_t* __restrict__ out,
                              int K, int N) {
  __shared__ float tile[32][33];
  const int tx = threadIdx.x, ty = threadIdx.y;
  const int n0 = blockIdx.x * 32;
  const int k0 = blockIdx.y * 32;
#pragma unroll
  for (int i = 0; i < 4; ++i) {
    const int k = k0 + ty + i * 8;
    tile[ty + i * 8][tx] = in[(size_t)k * N + (n0 + tx)];
  }
  __syncthreads();
#pragma unroll
  for (int i = 0; i < 4; ++i) {
    const int n = n0 + ty + i * 8;
    out[(size_t)n * K + (k0 + tx)] = (bf16_t)tile[tx][ty + i * 8];
  }
}

// ---------------------------------------------------------------------------
// 3) GEMM: C[M,N] = A[M,K] * BT[N,K]^T, bf16 WMMA, TDM double-buffered LDS.
//    Block tile 128x128, 256 threads = 8 waves (4Mx2N); wave 0 drives the TDM.
// ---------------------------------------------------------------------------
template <typename OutT>
__global__ __launch_bounds__(256)
void gemm_bf16_wmma(const bf16_t* __restrict__ A, const bf16_t* __restrict__ BT,
                    OutT* __restrict__ C, int M, int N, int K) {
  __shared__ __align__(32) bf16_t Asm[2][128][32];
  __shared__ __align__(32) bf16_t Bsm[2][128][32];

  const int tid  = threadIdx.x;
  const int lane = tid & 31;
  const int w    = tid >> 5;
  const int wm   = w & 3;       // M offset *32
  const int wn   = w >> 2;      // N offset *64
  const int m0   = blockIdx.y * 128;
  const int n0   = blockIdx.x * 128;

  const int hl = (lane >> 4) & 1;   // half-wave select
  const int ln = lane & 15;

  v8f acc[2][4];
#pragma unroll
  for (int mi = 0; mi < 2; ++mi)
#pragma unroll
    for (int ni = 0; ni < 4; ++ni) acc[mi][ni] = (v8f){};

  const int nsteps = K / 32;
  // Prologue: stage tile 0 into buffer 0.
  if (w == 0) {
    tdm_load_2d(lds_off(&Asm[0][0][0]), A + (size_t)m0 * K, 32u, 128u, (unsigned long long)K);
    tdm_load_2d(lds_off(&Bsm[0][0][0]), BT + (size_t)n0 * K, 32u, 128u, (unsigned long long)K);
  }

  for (int i = 0; i < nsteps; ++i) {
    const int buf = i & 1;
    __syncthreads();                    // everyone done reading buf^1 (prev iter)
    if (w == 0) {
      if (i + 1 < nsteps) {
        const int k1 = (i + 1) * 32;
        tdm_load_2d(lds_off(&Asm[buf ^ 1][0][0]), A + (size_t)m0 * K + k1, 32u, 128u,
                    (unsigned long long)K);
        tdm_load_2d(lds_off(&Bsm[buf ^ 1][0][0]), BT + (size_t)n0 * K + k1, 32u, 128u,
                    (unsigned long long)K);
        __builtin_amdgcn_s_wait_tensorcnt((unsigned short)2);  // tile i landed (in-order)
      } else {
        __builtin_amdgcn_s_wait_tensorcnt((unsigned short)0);
      }
    }
    __syncthreads();                    // tile i visible to all waves

    // A fragment: lanes<16 hold K {0..15}, lanes>=16 hold {16..31}
    v16bf af[2], bfrag[4];
    const int klo = hl * 16;
#pragma unroll
    for (int mi = 0; mi < 2; ++mi) {
      const int r = wm * 32 + mi * 16 + ln;
      af[mi] = pack16(*(const v8bf*)&Asm[buf][r][klo], *(const v8bf*)&Asm[buf][r][klo + 8]);
    }
    const int kb = hl * 16;
#pragma unroll
    for (int ni = 0; ni < 4; ++ni) {
      const int n = wn * 64 + ni * 16 + ln;
      bfrag[ni] = pack16(*(const v8bf*)&Bsm[buf][n][kb], *(const v8bf*)&Bsm[buf][n][kb + 8]);
    }
#pragma unroll
    for (int mi = 0; mi < 2; ++mi)
#pragma unroll
      for (int ni = 0; ni < 4; ++ni)
        acc[mi][ni] = __builtin_amdgcn_wmma_f32_16x16x32_bf16(
            false, af[mi], false, bfrag[ni], (short)0, acc[mi][ni], false, false);
  }

  // Epilogue: C layout VGPR r -> row M = r + 8*hl, col N = ln
#pragma unroll
  for (int mi = 0; mi < 2; ++mi)
#pragma unroll
    for (int ni = 0; ni < 4; ++ni)
#pragma unroll
      for (int r = 0; r < 8; ++r) {
        const int row = m0 + wm * 32 + mi * 16 + r + hl * 8;
        const int col = n0 + wn * 64 + ni * 16 + ln;
        C[(size_t)row * N + col] = (OutT)acc[mi][ni][r];
      }
}

// ---------------------------------------------------------------------------
// 4) Flash attention, causal. qkv bf16 [B*T, 6144]. One block per (b,h,128 q).
//    K tile staged via TDM (overlapped with manual transposed V staging).
// ---------------------------------------------------------------------------
__global__ __launch_bounds__(256)
void flash_attn_wmma(const bf16_t* __restrict__ qkv, bf16_t* __restrict__ y) {
  __shared__ __align__(32) bf16_t Ksm[32][128];     // 32 keys x 128 d (TDM dest)
  __shared__ __align__(32) bf16_t Vt[128][32];      // transposed: d x keys
  __shared__ __align__(32) bf16_t Psm[8][16][32];   // per-wave P scratch

  const int tid  = threadIdx.x;
  const int lane = tid & 31;
  const int w    = tid >> 5;
  const int hl   = (lane >> 4) & 1;
  const int ln   = lane & 15;

  const int b  = blockIdx.y >> 4;
  const int h  = blockIdx.y & 15;
  const int q0 = blockIdx.x * 128;
  const int qw = q0 + w * 16;                       // wave's first q row

  const float scale = 0.08838834764831845f;         // 1/sqrt(128)

  // Q A-fragments over d (4 x 16x32), kept in VGPRs for the whole pass
  v16bf qa[4];
  {
    const int t = qw + ln;
    const bf16_t* qp = qkv + (size_t)(b * SEQ_T + t) * QKV_N + h * HEAD_DIM;
    const int klo = hl * 16;
#pragma unroll
    for (int dk = 0; dk < 4; ++dk)
      qa[dk] = pack16(*(const v8bf*)(qp + dk * 32 + klo),
                      *(const v8bf*)(qp + dk * 32 + klo + 8));
  }

  v8f acc[8];
#pragma unroll
  for (int i = 0; i < 8; ++i) acc[i] = (v8f){};
  float mrow[8], lsum[8];
#pragma unroll
  for (int r = 0; r < 8; ++r) { mrow[r] = -1e30f; lsum[r] = 0.0f; }

  const int nblk  = q0 / 32 + 4;       // key blocks covering keys 0..q0+127
  const int skrow = tid >> 3;          // staging: key row 0..31
  const int sdcol = (tid & 7) * 16;    // staging: d chunk

  for (int j = 0; j < nblk; ++j) {
    const int kg0 = j * 32;
    __syncthreads();
    // K tile via TDM: 32 rows x 128 contiguous bf16, row stride 6144.
    if (w == 0)
      tdm_load_2d(lds_off(&Ksm[0][0]),
                  qkv + (size_t)(b * SEQ_T + kg0) * QKV_N + h * HEAD_DIM + D_MODEL,
                  128u, 32u, (unsigned long long)QKV_N);
    { // V staged manually with transpose (TDM has no transpose mode)
      const bf16_t* vp = qkv + (size_t)(b * SEQ_T + kg0 + skrow) * QKV_N
                             + h * HEAD_DIM + 2 * D_MODEL + sdcol;
      const v8bf* vv = (const v8bf*)vp;
      v8bf vlo = vv[0], vhi = vv[1];
#pragma unroll
      for (int i = 0; i < 8; ++i) {
        Vt[sdcol + i][skrow]     = vlo[i];
        Vt[sdcol + 8 + i][skrow] = vhi[i];
      }
    }
    if (w == 0) __builtin_amdgcn_s_wait_tensorcnt((unsigned short)0);
    __syncthreads();

    if (kg0 > qw + 15) continue;       // block fully above diagonal (wave-uniform)

    // S = Q K^T (16 q x 32 keys), accumulate over 4 d-chunks
    v8f s0 = (v8f){}, s1 = (v8f){};
    const int kb = hl * 16;
#pragma unroll
    for (int dk = 0; dk < 4; ++dk) {
      v16bf b0 = pack16(*(const v8bf*)&Ksm[ln][dk * 32 + kb],
                        *(const v8bf*)&Ksm[ln][dk * 32 + kb + 8]);
      s0 = __builtin_amdgcn_wmma_f32_16x16x32_bf16(false, qa[dk], false, b0, (short)0, s0, false, false);
      v16bf b1 = pack16(*(const v8bf*)&Ksm[16 + ln][dk * 32 + kb],
                        *(const v8bf*)&Ksm[16 + ln][dk * 32 + kb + 8]);
      s1 = __builtin_amdgcn_wmma_f32_16x16x32_bf16(false, qa[dk], false, b1, (short)0, s1, false, false);
    }

    // scale + causal mask + online softmax (rows live in 16-lane groups)
    const int c0 = kg0 + ln, c1 = c0 + 16;
    float p0[8], p1[8];
#pragma unroll
    for (int r = 0; r < 8; ++r) {
      const int row = qw + r + hl * 8;
      float v0 = s0[r] * scale;
      float v1 = s1[r] * scale;
      if (c0 > row) v0 = -1e30f;
      if (c1 > row) v1 = -1e30f;
      float mx = fmaxf(v0, v1);
#pragma unroll
      for (int off = 8; off >= 1; off >>= 1) mx = fmaxf(mx, __shfl_xor(mx, off));
      const float mnew = fmaxf(mrow[r], mx);
      const float sc   = __expf(mrow[r] - mnew);
      const float e0   = __expf(v0 - mnew);
      const float e1   = __expf(v1 - mnew);
      float rs = e0 + e1;
#pragma unroll
      for (int off = 8; off >= 1; off >>= 1) rs += __shfl_xor(rs, off);
      lsum[r] = lsum[r] * sc + rs;
      mrow[r] = mnew;
      p0[r] = e0; p1[r] = e1;
#pragma unroll
      for (int dn = 0; dn < 8; ++dn) acc[dn][r] *= sc;
    }

    // C-layout P -> LDS -> A-fragment layout (wave-private, DS in-order per wave)
#pragma unroll
    for (int r = 0; r < 8; ++r) {
      const int row = r + hl * 8;
      Psm[w][row][ln]      = (bf16_t)p0[r];
      Psm[w][row][16 + ln] = (bf16_t)p1[r];
    }
    const int klo = hl * 16;
    v16bf pf = pack16(*(const v8bf*)&Psm[w][ln][klo],
                      *(const v8bf*)&Psm[w][ln][klo + 8]);

    // acc += P(16x32) @ V(32x128)
#pragma unroll
    for (int dn = 0; dn < 8; ++dn) {
      const int d = dn * 16 + ln;
      v16bf vb = pack16(*(const v8bf*)&Vt[d][kb], *(const v8bf*)&Vt[d][kb + 8]);
      acc[dn] = __builtin_amdgcn_wmma_f32_16x16x32_bf16(false, pf, false, vb, (short)0, acc[dn], false, false);
    }
  }

  // normalize and store y bf16 [B*T, D_MODEL]
#pragma unroll
  for (int r = 0; r < 8; ++r) {
    const float linv = 1.0f / lsum[r];
    const int t = qw + r + hl * 8;
    bf16_t* yp = y + (size_t)(b * SEQ_T + t) * D_MODEL + h * HEAD_DIM;
#pragma unroll
    for (int dn = 0; dn < 8; ++dn)
      yp[dn * 16 + ln] = (bf16_t)(acc[dn][r] * linv);
  }
}

// ---------------------------------------------------------------------------
extern "C" void kernel_launch(void* const* d_in, const int* in_sizes, int n_in,
                              void* d_out, int out_size, void* d_ws, size_t ws_size,
                              hipStream_t stream) {
  (void)in_sizes; (void)n_in; (void)out_size; (void)ws_size;
  const float* x      = (const float*)d_in[0];   // [B,T,2048]
  const float* w_qkv  = (const float*)d_in[1];   // [2048,6144]
  const float* w_proj = (const float*)d_in[2];   // [2048,2048]
  float* out = (float*)d_out;                    // [B,T,2048] fp32

  char* ws = (char*)d_ws;
  bf16_t* xb     = (bf16_t*)ws; ws += (size_t)NROW    * D_MODEL * 2;
  bf16_t* wqkvT  = (bf16_t*)ws; ws += (size_t)QKV_N   * D_MODEL * 2;
  bf16_t* wprojT = (bf16_t*)ws; ws += (size_t)D_MODEL * D_MODEL * 2;
  bf16_t* qkvb   = (bf16_t*)ws; ws += (size_t)NROW    * QKV_N   * 2;
  bf16_t* yb     = (bf16_t*)ws;

  { // x -> bf16
    const int n = NROW * D_MODEL;
    cvt_f32_bf16<<<n / 4 / 256, 256, 0, stream>>>(x, xb, n);
  }
  // weights -> bf16, transposed to [N,K] so B-fragments load contiguously
  transpose_cvt<<<dim3(QKV_N / 32, D_MODEL / 32), dim3(32, 8), 0, stream>>>(w_qkv, wqkvT, D_MODEL, QKV_N);
  transpose_cvt<<<dim3(D_MODEL / 32, D_MODEL / 32), dim3(32, 8), 0, stream>>>(w_proj, wprojT, D_MODEL, D_MODEL);

  // QKV projection: [4096,2048] x [2048,6144] -> bf16 [4096,6144]
  gemm_bf16_wmma<bf16_t><<<dim3(QKV_N / 128, NROW / 128), 256, 0, stream>>>(
      xb, wqkvT, qkvb, NROW, QKV_N, D_MODEL);

  // causal flash attention -> y bf16 [4096,2048]
  flash_attn_wmma<<<dim3(SEQ_T / 128, BATCH * N_HEAD), 256, 0, stream>>>(qkvb, yb);

  // output projection: [4096,2048] x [2048,2048] -> fp32 out
  gemm_bf16_wmma<float><<<dim3(D_MODEL / 128, NROW / 128), 256, 0, stream>>>(
      yb, wprojT, out, NROW, D_MODEL, D_MODEL);
}
